// ClusterForecasting_71794673320055
// MI455X (gfx1250) — matrix operand order repaired and checked by reading, not compiled, not run on hardware
//
#include <hip/hip_runtime.h>
#include <math.h>

typedef __attribute__((ext_vector_type(16))) _Float16 v16h;
typedef __attribute__((ext_vector_type(8)))  float    v8f;

#define K_CLUSTERS 5
#define L_SNIP 8
#define F_IN 32
#define D_MODEL 64
#define P_DIM 512          // L * D
#define H_STRIDE 72        // padded LDS row stride (halves) for h tile
#define GN 40              // G-matrix columns: K_CLUSTERS * L_SNIP
#define GN_PAD 48          // padded to 3 WMMA n-tiles
#define WPB 8              // waves per block (wave32)
#define TPB 256

// ---------------- init: zero the 10-float accumulator workspace --------------
__global__ void cf_init_ws(float* ws) {
  if (threadIdx.x < 16) ws[threadIdx.x] = 0.0f;
}

// ---------------- fused GEMM1 + GEMM2 + WMMA cluster-distance pass -----------
__global__ __launch_bounds__(TPB)
void cf_main(const float* __restrict__ x_seg,
             const float* __restrict__ W_emb, const float* __restrict__ b_emb,
             const float* __restrict__ W_seq, const float* __restrict__ b_seq,
             const float* __restrict__ cent,
             float* __restrict__ seg_sum, float* __restrict__ seg_cnt,
             int n_pairs) {
  __shared__ _Float16 sWemb[F_IN * D_MODEL];        // 4 KB
  __shared__ _Float16 sWseq[D_MODEL * D_MODEL];     // 8 KB
  __shared__ _Float16 sGm[D_MODEL * GN_PAD];        // 6 KB: G[d1][c], c=k*8+r
  __shared__ float    sCnorm[K_CLUSTERS];
  __shared__ float    sBC[K_CLUSTERS];              // bias-correction dots
  __shared__ float    sB1[D_MODEL];
  __shared__ float    sB2[D_MODEL];
  __shared__ _Float16 sH[WPB][16 * H_STRIDE];       // 18 KB, per-wave h tiles
  __shared__ float    sSum[K_CLUSTERS];
  __shared__ float    sCnt[K_CLUSTERS];

  const int tid  = threadIdx.x;
  const int lane = tid & 31;
  const int wave = tid >> 5;
  const int nc   = lane & 15;   // column index within a 16-wide tile (B/C/D frags)
  const int half = lane >> 4;   // 0: lanes 0-15, 1: lanes 16-31
  const int klo  = half ? 8 : 0;

  // ---- cooperative LDS fill ----
  for (int i = tid; i < F_IN * D_MODEL; i += TPB)    sWemb[i] = (_Float16)W_emb[i];
  for (int i = tid; i < D_MODEL * D_MODEL; i += TPB) sWseq[i] = (_Float16)W_seq[i];
  for (int i = tid; i < D_MODEL; i += TPB) { sB1[i] = b_emb[i]; sB2[i] = b_seq[i]; }

  // G[d1][c] = sum_d2 W_seq[d1][d2] * cent[k*512 + r*64 + d2], c = k*8 + r
  for (int idx = tid; idx < D_MODEL * GN_PAD; idx += TPB) {
    int c  = idx % GN_PAD;
    int d1 = idx / GN_PAD;
    float s = 0.0f;
    if (c < GN) {
      int k = c >> 3, r = c & 7;
      const float* wrow = W_seq + (size_t)d1 * D_MODEL;
      const float* crow = cent + (size_t)k * P_DIM + r * D_MODEL;
      for (int d2 = 0; d2 < D_MODEL; ++d2) s = fmaf(wrow[d2], crow[d2], s);
    }
    sGm[idx] = (_Float16)s;
  }
  if (tid < K_CLUSTERS) {   // centroid norms
    float s = 0.0f;
    for (int i = 0; i < P_DIM; ++i) { float c = cent[tid * P_DIM + i]; s = fmaf(c, c, s); }
    sCnorm[tid] = s;
    sSum[tid] = 0.0f;
    sCnt[tid] = 0.0f;
  }
  if (tid >= 32 && tid < 32 + K_CLUSTERS) {  // bc[k] = sum_{r,d} b_seq[d]*cent[k][r*64+d]
    int k = tid - 32;
    float s = 0.0f;
    for (int i = 0; i < P_DIM; ++i)
      s = fmaf(b_seq[i & (D_MODEL - 1)], cent[(size_t)k * P_DIM + i], s);
    sBC[k] = s;
  }
  __syncthreads();

  // ---- per-wave persistent B fragments ----
  // B layout: lanes 0-15 hold K=0..7,16..23 across the 16 halves, lanes 16-31 hold K=8..15,24..31.
  v16h bemb[4];      // W_emb 32x64 : 4 n-tiles, K=32
  v16h bseq[2][4];   // W_seq 64x64 : 2 k-steps x 4 n-tiles
  v16h bg[2][3];     // G     64x48 : 2 k-steps x 3 n-tiles
#pragma unroll
  for (int t = 0; t < 4; ++t) {
#pragma unroll
    for (int i = 0; i < 16; ++i) {
      int k = (i < 8 ? i : i + 8) + 8 * half;
      bemb[t][i] = sWemb[k * D_MODEL + t * 16 + nc];
    }
#pragma unroll
    for (int ks = 0; ks < 2; ++ks)
#pragma unroll
      for (int i = 0; i < 16; ++i) {
        int k = ks * 32 + (i < 8 ? i : i + 8) + 8 * half;
        bseq[ks][t][i] = sWseq[k * D_MODEL + t * 16 + nc];
      }
  }
#pragma unroll
  for (int ks = 0; ks < 2; ++ks)
#pragma unroll
    for (int t = 0; t < 3; ++t)
#pragma unroll
      for (int i = 0; i < 16; ++i) {
        int k = ks * 32 + (i < 8 ? i : i + 8) + 8 * half;
        bg[ks][t][i] = sGm[k * GN_PAD + t * 16 + nc];
      }

  const int gwave  = blockIdx.x * WPB + wave;
  const int nwaves = gridDim.x * WPB;
  const int m = nc;  // A-fragment row within the 16-row tile

  // ---- software pipeline: raw f32 A-tile buffered one iteration ahead ----
  float raw[16];
  auto tile_ptr = [&](int p) -> const float* {
    return x_seg + ((size_t)p * 16 + m) * F_IN;
  };
  auto load_raw = [&](int p) {
    const float* rp = tile_ptr(p);
#pragma unroll
    for (int i = 0; i < 8; ++i) raw[i]     = rp[klo + i];
#pragma unroll
    for (int i = 0; i < 8; ++i) raw[8 + i] = rp[klo + 16 + i];
  };

  if (gwave < n_pairs) load_raw(gwave);

  for (int pair = gwave; pair < n_pairs; pair += nwaves) {
    const int nxt = pair + nwaves;
    // prefetch two tiles ahead toward L2 (global_prefetch_b8)
    if (nxt + nwaves < n_pairs)
      __builtin_prefetch(tile_ptr(nxt + nwaves) + klo, 0, 1);

    // ---- A1 fragment: consume buffered raw f32 -> f16 ----
    v16h a1;
#pragma unroll
    for (int i = 0; i < 16; ++i) a1[i] = (_Float16)raw[i];

    // issue next tile's global loads now; latency overlaps GEMM2/GEMM3 below
    if (nxt < n_pairs) load_raw(nxt);

    // ---- GEMM1: h = x_seg @ W_emb (4 WMMAs, K=32 exact) ----
    v8f acc1[4];
#pragma unroll
    for (int t = 0; t < 4; ++t) {
      v8f z = {};
      acc1[t] = __builtin_amdgcn_wmma_f32_16x16x32_f16(
          false, a1, false, bemb[t], (short)0, z, false, false);
    }

    // ---- bias + transpose through per-wave LDS tile (col-frag -> row-frag) ----
    _Float16* hbase = &sH[wave][0];
#pragma unroll
    for (int t = 0; t < 4; ++t) {
      float bias = sB1[t * 16 + nc];
#pragma unroll
      for (int j = 0; j < 8; ++j) {
        float v = acc1[t][j] + bias;
        hbase[(j + 8 * half) * H_STRIDE + t * 16 + nc] = (_Float16)v;
      }
    }

    // ---- GEMM2 (out) + GEMM3 (centroid dots): 2 k-steps x (4+3) WMMAs ----
    v8f acc2[4];
    v8f part[3];
    {
      v8f z = {};
#pragma unroll
      for (int t = 0; t < 4; ++t) acc2[t] = z;
#pragma unroll
      for (int t = 0; t < 3; ++t) part[t] = z;
    }
#pragma unroll
    for (int ks = 0; ks < 2; ++ks) {
      const _Float16* hr = &sH[wave][m * H_STRIDE + ks * 32];
      v16h a2;
#pragma unroll
      for (int i = 0; i < 8; ++i) a2[i]     = hr[klo + i];
#pragma unroll
      for (int i = 0; i < 8; ++i) a2[8 + i] = hr[klo + 16 + i];
#pragma unroll
      for (int t = 0; t < 4; ++t)
        acc2[t] = __builtin_amdgcn_wmma_f32_16x16x32_f16(
            false, a2, false, bseq[ks][t], (short)0, acc2[t], false, false);
#pragma unroll
      for (int t = 0; t < 3; ++t)
        part[t] = __builtin_amdgcn_wmma_f32_16x16x32_f16(
            false, a2, false, bg[ks][t], (short)0, part[t], false, false);
    }

    // ---- point norm (VALU, 32 FMAs/lane) ----
    float nrm = 0.0f;
#pragma unroll
    for (int t = 0; t < 4; ++t) {
      float bias = sB2[t * 16 + nc];
#pragma unroll
      for (int j = 0; j < 8; ++j) {
        float v = acc2[t][j] + bias;
        nrm = fmaf(v, v, nrm);
      }
    }
#pragma unroll
    for (int off = 1; off < 16; off <<= 1) nrm += __shfl_xor(nrm, off, 32);

    // ---- diagonal extraction of the dot GEMM ----
    // partial[M=j+8*half][c=t*16+nc]; cluster k=2t+g terms live at lanes
    // nc = g*8 + j (g = nc>>3), element index j = nc&7.
    const int jj = nc & 7;
    float s0 = part[0][jj];
    float s1 = part[1][jj];
    float s2 = part[2][jj];
#pragma unroll
    for (int off = 1; off < 8; off <<= 1) {
      s0 += __shfl_xor(s0, off, 32);
      s1 += __shfl_xor(s1, off, 32);
      s2 += __shfl_xor(s2, off, 32);
    }
    // group g=0 lanes hold k={0,2,4}; pull k={1,3} from group g=1 (lane half*16+8)
    const int src8 = (half << 4) + 8;
    float d1v = __shfl(s0, src8, 32);
    float d3v = __shfl(s1, src8, 32);

    if (nc == 0) {  // lane 0 -> point 2*pair, lane 16 -> point 2*pair+1
      float dots[K_CLUSTERS];
      dots[0] = s0  + sBC[0];
      dots[1] = d1v + sBC[1];
      dots[2] = s1  + sBC[2];
      dots[3] = d3v + sBC[3];
      dots[4] = s2  + sBC[4];
      float best = nrm - 2.0f * dots[0] + sCnorm[0];
      int bi = 0;
#pragma unroll
      for (int k = 1; k < K_CLUSTERS; ++k) {
        float d2 = nrm - 2.0f * dots[k] + sCnorm[k];
        if (d2 < best) { best = d2; bi = k; }
      }
      atomicAdd(&sSum[bi], best);   // ds_add_f32
      atomicAdd(&sCnt[bi], 1.0f);
    }
  }

  __syncthreads();
  if (tid < K_CLUSTERS) {
    atomicAdd(&seg_sum[tid], sSum[tid]);  // 10 global atomics per block total
    atomicAdd(&seg_cnt[tid], sCnt[tid]);
  }
}

// ---------------- finalize: fold 10 floats into the scalar loss --------------
__global__ void cf_finalize(const float* __restrict__ ws,
                            const float* __restrict__ w_loss,
                            float* __restrict__ out, float n_points) {
  if (threadIdx.x == 0 && blockIdx.x == 0) {
    float sums[K_CLUSTERS], cnts[K_CLUSTERS];
    float total = 0.0f, ctot = 0.0f;
#pragma unroll
    for (int k = 0; k < K_CLUSTERS; ++k) {
      sums[k] = ws[k];
      cnts[k] = ws[K_CLUSTERS + k];
      total += sums[k];
      ctot  += cnts[k];
    }
    float inter = total / n_points;

    float intra_num = 0.0f, ne = 0.0f;
#pragma unroll
    for (int k = 0; k < K_CLUSTERS; ++k)
      if (cnts[k] > 0.0f) { intra_num += sums[k] / cnts[k]; ne += 1.0f; }
    float intra = (ne > 0.0f) ? (intra_num / ne) : 0.0f;

    // counts = [c0..c4, 0] (bincount length K+1); unbiased std (ddof=1)
    float mean = ctot / 6.0f;
    float var = mean * mean;  // trailing zero bin: (0 - mean)^2
#pragma unroll
    for (int k = 0; k < K_CLUSTERS; ++k) {
      float d = cnts[k] - mean;
      var = fmaf(d, d, var);
    }
    var /= 5.0f;
    float entropy = sqrtf(var);

    out[0] = w_loss[0] * inter + w_loss[1] * intra + w_loss[2] * entropy;
  }
}

extern "C" void kernel_launch(void* const* d_in, const int* in_sizes, int n_in,
                              void* d_out, int out_size, void* d_ws, size_t ws_size,
                              hipStream_t stream) {
  // inputs: 0:x (unused), 1:x_seg, 2:W_emb, 3:b_emb, 4:W_seq, 5:b_seq,
  //         6:centroids, 7:w_loss
  const float* x_seg = (const float*)d_in[1];
  const float* W_emb = (const float*)d_in[2];
  const float* b_emb = (const float*)d_in[3];
  const float* W_seq = (const float*)d_in[4];
  const float* b_seq = (const float*)d_in[5];
  const float* cent  = (const float*)d_in[6];
  const float* wls   = (const float*)d_in[7];
  float* ws = (float*)d_ws;  // [0..4]=seg_sum, [5..9]=seg_cnt

  int rows     = in_sizes[1] / F_IN;   // B*S*L
  int n_points = rows / L_SNIP;        // B*S
  int n_pairs  = n_points / 2;         // 16-row tiles

  cf_init_ws<<<1, 32, 0, stream>>>(ws);

  int blocks = (n_pairs + WPB - 1) / WPB;
  if (blocks > 1024) blocks = 1024;
  if (blocks < 1) blocks = 1;
  cf_main<<<blocks, TPB, 0, stream>>>(x_seg, W_emb, b_emb, W_seq, b_seq, cent,
                                      ws, ws + K_CLUSTERS, n_pairs);

  cf_finalize<<<1, 32, 0, stream>>>(ws, wls, (float*)d_out, (float)n_points);
}
